// AsyncNaive_1297080124025
// MI455X (gfx1250) — matrix operand order repaired and checked by reading, not compiled, not run on hardware
//
#include <hip/hip_runtime.h>
#include <stdint.h>

#define T_STEPS 256
#define K_STEPS 4
#define N_BATCH 64
#define R_ROWS (N_BATCH * T_STEPS * K_STEPS)   // 65536
#define TOTAL_HW 492

// ---------- WMMA types ----------
typedef __bf16 bf16x16 __attribute__((ext_vector_type(16)));
typedef float  f32x8   __attribute__((ext_vector_type(8)));

static __device__ __forceinline__ unsigned short f2bf(float f) {
  unsigned u = __float_as_uint(f);
  u += 0x7fffu + ((u >> 16) & 1u);   // round-to-nearest-even
  return (unsigned short)(u >> 16);
}
static __device__ __forceinline__ float bf2f(unsigned short h) {
  return __uint_as_float(((unsigned)h) << 16);
}
static __device__ __forceinline__ bf16x16 mkfrag(uint4 lo, uint4 hi) {
  union { uint4 q[2]; bf16x16 v; } u;
  u.q[0] = lo; u.q[1] = hi;
  return u.v;
}
static __device__ __forceinline__ f32x8 wmma_bf16(bf16x16 a, bf16x16 b, f32x8 c) {
  return __builtin_amdgcn_wmma_f32_16x16x32_bf16(false, a, false, b, (short)0, c,
                                                 false, false);
}
static __device__ __forceinline__ float sigmoidf_(float x) {
  return 1.0f / (1.0f + __expf(-x));
}

// ---------- packing kernels ----------
// Gate-aligned, zero-padded bf16 weight: out[(g*Hp+j)*Dp + d] = W[(g*H+j)*D + d]
__global__ void pack_w_kernel(const float* __restrict__ W, unsigned short* __restrict__ out,
                              int H, int Hp, int D, int Dp) {
  size_t total = (size_t)4 * Hp * Dp;
  for (size_t idx = (size_t)blockIdx.x * blockDim.x + threadIdx.x; idx < total;
       idx += (size_t)gridDim.x * blockDim.x) {
    size_t row = idx / Dp; int d = (int)(idx % Dp);
    int g = (int)(row / Hp), j = (int)(row % Hp);
    float v = (j < H && d < D) ? W[((size_t)(g * H + j)) * D + d] : 0.0f;
    out[idx] = f2bf(v);
  }
}

__global__ void pack_bias_kernel(const float* __restrict__ bih, const float* __restrict__ bhh,
                                 float* __restrict__ out, int H, int Hp) {
  int total = 4 * Hp;
  for (int idx = blockIdx.x * blockDim.x + threadIdx.x; idx < total;
       idx += gridDim.x * blockDim.x) {
    int g = idx / Hp, j = idx % Hp;
    out[idx] = (j < H) ? (bih[g * H + j] + bhh[g * H + j]) : 0.0f;
  }
}

__global__ void pack_x_kernel(const float* __restrict__ x, unsigned short* __restrict__ out,
                              int D, int Dp) {
  size_t total = (size_t)R_ROWS * Dp;
  for (size_t idx = (size_t)blockIdx.x * blockDim.x + threadIdx.x; idx < total;
       idx += (size_t)gridDim.x * blockDim.x) {
    size_t r = idx / Dp; int d = (int)(idx % Dp);
    out[idx] = (d < D) ? f2bf(x[r * D + d]) : (unsigned short)0;
  }
}

// ---------- phase A: G = Xb @ Wih'^T + bias', bf16 out in (tk, j, n) layout ----------
// 2x4 register blocking: each wave computes a 32x64 output macro-tile (8 WMMA tiles),
// sharing 2 A-fragments and 4 B-fragments per K-step (1.5 loads per wmma).
__global__ __launch_bounds__(256) void gemm_g_kernel(const unsigned short* __restrict__ Xb,
                                                     const unsigned short* __restrict__ Wp,
                                                     const float* __restrict__ bias,
                                                     unsigned short* __restrict__ G,
                                                     int Dp, int FH) {
  const int colGroups = FH >> 6;                       // 64-wide column groups
  const long totalUnits = (long)(R_ROWS / 32) * colGroups;
  const int lane = threadIdx.x & 31;
  const int nl = lane & 15, hi = lane >> 4;
  long gw = ((long)blockIdx.x * blockDim.x + threadIdx.x) >> 5;
  long stride = ((long)gridDim.x * blockDim.x) >> 5;
  for (long unit = gw; unit < totalUnits; unit += stride) {
    const int rowG = (int)(unit / colGroups);
    const int colG = (int)(unit % colGroups);
    const int r0 = rowG * 32;
    const int c0 = colG * 64;
    f32x8 acc[2][4];
    const f32x8 zero = {0.f, 0.f, 0.f, 0.f, 0.f, 0.f, 0.f, 0.f};
#pragma unroll
    for (int rr = 0; rr < 2; ++rr)
#pragma unroll
      for (int cc = 0; cc < 4; ++cc) acc[rr][cc] = zero;

    const unsigned short* arow0 = Xb + (size_t)(r0 + nl) * Dp;        // A rows (M=lane&15)
    const unsigned short* arow1 = Xb + (size_t)(r0 + 16 + nl) * Dp;
    for (int k0 = 0; k0 < Dp; k0 += 32) {
      const unsigned short* ap0 = arow0 + k0 + hi * 8;
      const unsigned short* ap1 = arow1 + k0 + hi * 8;
      bf16x16 a0 = mkfrag(*(const uint4*)ap0, *(const uint4*)(ap0 + 16));
      bf16x16 a1 = mkfrag(*(const uint4*)ap1, *(const uint4*)(ap1 + 16));
#pragma unroll
      for (int cc = 0; cc < 4; ++cc) {
        const unsigned short* bp =
            Wp + (size_t)(c0 + cc * 16 + nl) * Dp + k0 + hi * 8;      // B col (N=lane&15)
        bf16x16 b = mkfrag(*(const uint4*)bp, *(const uint4*)(bp + 16));
        acc[0][cc] = wmma_bf16(a0, b, acc[0][cc]);
        acc[1][cc] = wmma_bf16(a1, b, acc[1][cc]);
      }
    }
#pragma unroll
    for (int cc = 0; cc < 4; ++cc) {
      const int j = c0 + cc * 16 + nl;
      const float bv = bias[j];
#pragma unroll
      for (int rr = 0; rr < 2; ++rr) {
#pragma unroll
        for (int v = 0; v < 8; ++v) {
          int r = r0 + rr * 16 + hi * 8 + v;    // C: M = (lane/16)*8 + v
          int n = r >> 10;                      // r / (T*K)
          int tk = r & 1023;
          G[((size_t)tk * FH + j) * 64 + n] = f2bf(acc[rr][cc][v] + bv);
        }
      }
    }
  }
}

// ---------- phase B: sequential LSTM recurrence, one workgroup per modality ----------
struct ModP {
  const unsigned short* G;     // (1024, FH, 64) bf16
  const unsigned short* Whh;   // (FH, Hp) bf16, gate-aligned, K zero-padded
  const unsigned char*  mask;  // (64, T, K) bool
  float* hs;                   // (64, T, 492) f32, written at column offset `off`
  int Hp, H, FH, off;
};

__global__ __launch_bounds__(1024) void lstm_rec_kernel(ModP P0, ModP P1, ModP P2,
                                                        const int* __restrict__ seq) {
  __shared__ unsigned short h_lds[64 * 320];  // max Hp = 320 (linguistic), 40 KB
  ModP P = (blockIdx.x == 0) ? P0 : (blockIdx.x == 1) ? P1 : P2;
  const int Hp = P.Hp, FH = P.FH, H = P.H;
  const int tid = threadIdx.x, wave = tid >> 5, lane = tid & 31;
  const int nl = lane & 15, hi = lane >> 4;
  const int pairs = (Hp >> 4) * 4;  // (mt, jt) tile pairs; max 80

  for (int i = tid; i < 64 * Hp; i += blockDim.x) h_lds[i] = 0;

  float h_st[3][8], c_st[3][8];
#pragma unroll
  for (int i = 0; i < 3; ++i)
#pragma unroll
    for (int v = 0; v < 8; ++v) { h_st[i][v] = 0.f; c_st[i][v] = 0.f; }

  __syncthreads();

  for (int t = 0; t < T_STEPS; ++t) {
    for (int k = 0; k < K_STEPS; ++k) {
      const int tk = t * K_STEPS + k;
      // ---- GEMM z = G[tk] + h @ Whh'^T and cell update (registers only) ----
#pragma unroll
      for (int i = 0; i < 3; ++i) {
        const int pr = wave + i * 32;
        if (pr >= pairs) continue;           // uniform per wave
        const int mt = pr & 3, jt = pr >> 2;
        const int m0 = mt * 16;

        // prefetch next step's gate slab (G exceeds L2: hide HBM latency
        // behind this step's WMMA K-loop) -> global_prefetch_b8
        if (tk + 1 < T_STEPS * K_STEPS) {
#pragma unroll
          for (int g = 0; g < 4; ++g) {
            const unsigned short* npf =
                P.G + ((size_t)(tk + 1) * FH + (g * Hp + jt * 16 + nl)) * 64 + m0 + hi * 8;
            __builtin_prefetch((const void*)npf, 0, 1);
          }
        }

        f32x8 acc[4];
        const f32x8 zero = {0.f, 0.f, 0.f, 0.f, 0.f, 0.f, 0.f, 0.f};
#pragma unroll
        for (int g = 0; g < 4; ++g) acc[g] = zero;

        for (int k0 = 0; k0 < Hp; k0 += 32) {
          const unsigned short* ap = &h_lds[(m0 + nl) * Hp + k0 + hi * 8];
          bf16x16 a = mkfrag(*(const uint4*)ap, *(const uint4*)(ap + 16));
#pragma unroll
          for (int g = 0; g < 4; ++g) {
            const unsigned short* bp =
                P.Whh + (size_t)(g * Hp + jt * 16 + nl) * Hp + k0 + hi * 8;
            bf16x16 b = mkfrag(*(const uint4*)bp, *(const uint4*)(bp + 16));
            acc[g] = wmma_bf16(a, b, acc[g]);
          }
        }
        // add precomputed gates (one 16B load per gate: 8 contiguous batch elems)
        float z[4][8];
#pragma unroll
        for (int g = 0; g < 4; ++g) {
          const unsigned short* gp =
              P.G + ((size_t)tk * FH + (g * Hp + jt * 16 + nl)) * 64 + m0 + hi * 8;
          uint4 q = *(const uint4*)gp;
          unsigned w0 = q.x, w1 = q.y, w2 = q.z, w3 = q.w;
          z[g][0] = acc[g][0] + bf2f((unsigned short)(w0 & 0xffff));
          z[g][1] = acc[g][1] + bf2f((unsigned short)(w0 >> 16));
          z[g][2] = acc[g][2] + bf2f((unsigned short)(w1 & 0xffff));
          z[g][3] = acc[g][3] + bf2f((unsigned short)(w1 >> 16));
          z[g][4] = acc[g][4] + bf2f((unsigned short)(w2 & 0xffff));
          z[g][5] = acc[g][5] + bf2f((unsigned short)(w2 >> 16));
          z[g][6] = acc[g][6] + bf2f((unsigned short)(w3 & 0xffff));
          z[g][7] = acc[g][7] + bf2f((unsigned short)(w3 >> 16));
        }
#pragma unroll
        for (int v = 0; v < 8; ++v) {
          const int m = m0 + hi * 8 + v;
          const unsigned char mk = P.mask[((size_t)m * T_STEPS + t) * K_STEPS + k];
          float iv = sigmoidf_(z[0][v]);
          float fv = sigmoidf_(z[1][v]);
          float gv = tanhf(z[2][v]);
          float ov = sigmoidf_(z[3][v]);
          float c1 = fv * c_st[i][v] + iv * gv;
          float h1 = ov * tanhf(c1);
          if (mk) { c_st[i][v] = c1; h_st[i][v] = h1; }
        }
      }
      __syncthreads();  // all waves finished reading h_lds
      // ---- seq-length gating (end of t), write h back to LDS (+ hs at k==3) ----
#pragma unroll
      for (int i = 0; i < 3; ++i) {
        const int pr = wave + i * 32;
        if (pr >= pairs) continue;
        const int mt = pr & 3, jt = pr >> 2;
        const int m0 = mt * 16;
        const int j = jt * 16 + nl;
#pragma unroll
        for (int v = 0; v < 8; ++v) {
          const int m = m0 + hi * 8 + v;
          if (k == K_STEPS - 1) {
            if (t >= seq[m]) { h_st[i][v] = 0.f; c_st[i][v] = 0.f; }
            if (j < H)
              P.hs[((size_t)m * T_STEPS + t) * TOTAL_HW + P.off + j] = h_st[i][v];
          }
          h_lds[m * Hp + j] = f2bf(h_st[i][v]);
        }
      }
      __syncthreads();  // new h visible before next step's reads
    }
  }
}

// ---------- fuse head collapsed: w = W2@W1 (1x492), bias = W2.b1 + b2 ----------
__global__ void collapse_fuse_kernel(const float* __restrict__ W1, const float* __restrict__ b1,
                                     const float* __restrict__ W2, const float* __restrict__ b2,
                                     float* __restrict__ wf) {
  int d = blockIdx.x * blockDim.x + threadIdx.x;
  if (d < TOTAL_HW) {
    float s = 0.f;
    for (int j = 0; j < 256; ++j) s += W2[j] * W1[(size_t)j * TOTAL_HW + d];
    wf[d] = s;
  } else if (d == TOTAL_HW) {
    float s = b2[0];
    for (int j = 0; j < 256; ++j) s += W2[j] * b1[j];
    wf[TOTAL_HW] = s;
  }
}

__global__ void out_kernel(const float* __restrict__ hs, const float* __restrict__ wf,
                           const float* __restrict__ lm, float* __restrict__ out) {
  int idx = blockIdx.x * blockDim.x + threadIdx.x;
  if (idx >= N_BATCH * T_STEPS) return;
  const float* row = hs + (size_t)idx * TOTAL_HW;
  float acc = wf[TOTAL_HW];
  for (int d = 0; d < TOTAL_HW; ++d) acc += row[d] * wf[d];
  out[idx] = acc * lm[idx];
}

// ---------- workspace layout (bytes) ----------
#define SZ_XB_L ((size_t)R_ROWS * 320 * 2)
#define SZ_XB_A ((size_t)R_ROWS * 96 * 2)
#define SZ_XB_I ((size_t)R_ROWS * 128 * 2)
#define SZ_WIH_L ((size_t)1280 * 320 * 2)
#define SZ_WIH_A ((size_t)256 * 96 * 2)
#define SZ_WIH_I ((size_t)512 * 128 * 2)
#define SZ_WHH_L ((size_t)1280 * 320 * 2)
#define SZ_WHH_A ((size_t)256 * 64 * 2)
#define SZ_WHH_I ((size_t)512 * 128 * 2)
#define SZ_B_L ((size_t)1280 * 4)
#define SZ_B_A ((size_t)256 * 4)
#define SZ_B_I ((size_t)512 * 4)
#define SZ_G_L ((size_t)R_ROWS * 1280 * 2)
#define SZ_G_A ((size_t)R_ROWS * 256 * 2)
#define SZ_G_I ((size_t)R_ROWS * 512 * 2)
#define SZ_HS ((size_t)N_BATCH * T_STEPS * TOTAL_HW * 4)

#define OFF_XB_L ((size_t)0)
#define OFF_XB_A (OFF_XB_L + SZ_XB_L)
#define OFF_XB_I (OFF_XB_A + SZ_XB_A)
#define OFF_WIH_L (OFF_XB_I + SZ_XB_I)
#define OFF_WIH_A (OFF_WIH_L + SZ_WIH_L)
#define OFF_WIH_I (OFF_WIH_A + SZ_WIH_A)
#define OFF_WHH_L (OFF_WIH_I + SZ_WIH_I)
#define OFF_WHH_A (OFF_WHH_L + SZ_WHH_L)
#define OFF_WHH_I (OFF_WHH_A + SZ_WHH_A)
#define OFF_B_L (OFF_WHH_I + SZ_WHH_I)
#define OFF_B_A (OFF_B_L + SZ_B_L)
#define OFF_B_I (OFF_B_A + SZ_B_A)
#define OFF_G_L (OFF_B_I + SZ_B_I)
#define OFF_G_A (OFF_G_L + SZ_G_L)
#define OFF_G_I (OFF_G_A + SZ_G_A)
#define OFF_HS (OFF_G_I + SZ_G_I)
#define OFF_WF (OFF_HS + SZ_HS)

extern "C" void kernel_launch(void* const* d_in, const int* in_sizes, int n_in,
                              void* d_out, int out_size, void* d_ws, size_t ws_size,
                              hipStream_t stream) {
  (void)in_sizes; (void)n_in; (void)out_size; (void)ws_size;
  // inputs: per modality {x, mask, Wih, Whh, bih, bhh}, then seq, lstm_masks, fuse
  const float* xL = (const float*)d_in[0];
  const unsigned char* mL = (const unsigned char*)d_in[1];
  const float* WihL = (const float*)d_in[2];
  const float* WhhL = (const float*)d_in[3];
  const float* bihL = (const float*)d_in[4];
  const float* bhhL = (const float*)d_in[5];
  const float* xA = (const float*)d_in[6];
  const unsigned char* mA = (const unsigned char*)d_in[7];
  const float* WihA = (const float*)d_in[8];
  const float* WhhA = (const float*)d_in[9];
  const float* bihA = (const float*)d_in[10];
  const float* bhhA = (const float*)d_in[11];
  const float* xI = (const float*)d_in[12];
  const unsigned char* mI = (const unsigned char*)d_in[13];
  const float* WihI = (const float*)d_in[14];
  const float* WhhI = (const float*)d_in[15];
  const float* bihI = (const float*)d_in[16];
  const float* bhhI = (const float*)d_in[17];
  const int* seq = (const int*)d_in[18];
  const float* lm = (const float*)d_in[19];
  const float* fW1 = (const float*)d_in[20];
  const float* fb1 = (const float*)d_in[21];
  const float* fW2 = (const float*)d_in[22];
  const float* fb2 = (const float*)d_in[23];
  float* out = (float*)d_out;

  uint8_t* ws = (uint8_t*)d_ws;
  unsigned short* XbL = (unsigned short*)(ws + OFF_XB_L);
  unsigned short* XbA = (unsigned short*)(ws + OFF_XB_A);
  unsigned short* XbI = (unsigned short*)(ws + OFF_XB_I);
  unsigned short* pWihL = (unsigned short*)(ws + OFF_WIH_L);
  unsigned short* pWihA = (unsigned short*)(ws + OFF_WIH_A);
  unsigned short* pWihI = (unsigned short*)(ws + OFF_WIH_I);
  unsigned short* pWhhL = (unsigned short*)(ws + OFF_WHH_L);
  unsigned short* pWhhA = (unsigned short*)(ws + OFF_WHH_A);
  unsigned short* pWhhI = (unsigned short*)(ws + OFF_WHH_I);
  float* pBL = (float*)(ws + OFF_B_L);
  float* pBA = (float*)(ws + OFF_B_A);
  float* pBI = (float*)(ws + OFF_B_I);
  unsigned short* GL = (unsigned short*)(ws + OFF_G_L);
  unsigned short* GA = (unsigned short*)(ws + OFF_G_A);
  unsigned short* GI = (unsigned short*)(ws + OFF_G_I);
  float* hs = (float*)(ws + OFF_HS);
  float* wf = (float*)(ws + OFF_WF);

  // ---- pack weights / biases / inputs (bf16, gate-aligned, zero-padded) ----
  pack_w_kernel<<<1600, 256, 0, stream>>>(WihL, pWihL, 300, 320, 300, 320);
  pack_w_kernel<<<96, 256, 0, stream>>>(WihA, pWihA, 64, 64, 88, 96);
  pack_w_kernel<<<256, 256, 0, stream>>>(WihI, pWihI, 128, 128, 128, 128);
  pack_w_kernel<<<1600, 256, 0, stream>>>(WhhL, pWhhL, 300, 320, 300, 320);
  pack_w_kernel<<<64, 256, 0, stream>>>(WhhA, pWhhA, 64, 64, 64, 64);
  pack_w_kernel<<<256, 256, 0, stream>>>(WhhI, pWhhI, 128, 128, 128, 128);
  pack_bias_kernel<<<8, 256, 0, stream>>>(bihL, bhhL, pBL, 300, 320);
  pack_bias_kernel<<<1, 256, 0, stream>>>(bihA, bhhA, pBA, 64, 64);
  pack_bias_kernel<<<2, 256, 0, stream>>>(bihI, bhhI, pBI, 128, 128);
  pack_x_kernel<<<8192, 256, 0, stream>>>(xL, XbL, 300, 320);
  pack_x_kernel<<<4096, 256, 0, stream>>>(xA, XbA, 88, 96);
  pack_x_kernel<<<4096, 256, 0, stream>>>(xI, XbI, 128, 128);
  collapse_fuse_kernel<<<2, 256, 0, stream>>>(fW1, fb1, fW2, fb2, wf);

  // ---- phase A: WMMA GEMMs (2x4 macro-tiles per wave) ----
  gemm_g_kernel<<<5120, 256, 0, stream>>>(XbL, pWihL, pBL, GL, 320, 1280);
  gemm_g_kernel<<<1024, 256, 0, stream>>>(XbA, pWihA, pBA, GA, 96, 256);
  gemm_g_kernel<<<2048, 256, 0, stream>>>(XbI, pWihI, pBI, GI, 128, 512);

  // ---- phase B: persistent recurrence, one WGP-resident block per modality ----
  ModP PL = {GL, pWhhL, mL, hs, 320, 300, 1280, 0};
  ModP PA = {GA, pWhhA, mA, hs, 64, 64, 256, 300};
  ModP PI = {GI, pWhhI, mI, hs, 128, 128, 512, 364};
  lstm_rec_kernel<<<3, 1024, 0, stream>>>(PL, PA, PI, seq);

  // ---- fuse head (collapsed to a single dot per (n,t)) ----
  out_kernel<<<(N_BATCH * T_STEPS + 255) / 256, 256, 0, stream>>>(hs, wf, lm, out);
}
